// BallEncoder_6004364280406
// MI455X (gfx1250) — matrix operand order repaired and checked by reading, not compiled, not run on hardware
//
#include <hip/hip_runtime.h>

// ---------------------------------------------------------------------------
// Bidirectional masked LSTM (Keras semantics) for MI455X / gfx1250.
//
// Strategy (see analysis): weight-stationary recurrent GEMM in LDS, batch-
// parallel groups, per-step group sync (cluster barrier fast path + global
// atomic fallback), v_wmma_f32_16x16x32_bf16 for h @ Wh, f32 cell state.
//
// Grid: 256 blocks x 128 threads.
//   blockIdx.x = dir(1b) | btile(3b) | ublock(4b)
//   group = (dir, btile): 16 blocks cooperating on one batch tile of 16 rows.
//   Each block owns 32 units (=128 gate columns: i/f/g/o blocks).
//   Wave w (0..3) computes gate type w for all 32 units (2 WMMA N-tiles).
//
// Workspace: [0,4096) barrier counters (16 groups x 256B), then double-
// buffered bf16 h exchange buffer [2][2dir][128][512] = 512 KB.
// ---------------------------------------------------------------------------

#define UNITS      512
#define TSTEPS     512
#define BATCH      128
#define GATES      2048
#define BTILE      16
#define UBLK       32
#define NBLK_GROUP 16
#define THREADS    128

#define WS_CTR_STRIDE 256
#define WS_HS_OFF     4096

#define SH_OFF (128ull*512*1024)
#define SC_OFF (SH_OFF + 128ull*2*512)
#define PA_OFF (SC_OFF + 128ull*2*512)

// LDS: whfrag 128K + htile 16K + zbuf 8K + c 2K + h 2K + wx/b 2K + xt 192B
#define SMEM_BYTES (131072 + 16384 + 8192 + 2048 + 2048 + 2048 + 192)

typedef __attribute__((ext_vector_type(16))) __bf16          v16bf;
typedef __attribute__((ext_vector_type(8)))  float           v8f;
typedef __attribute__((ext_vector_type(8)))  unsigned short  ushort8;

union Frag {
  struct Halves { ushort8 lo, hi; } p;
  v16bf v;
};

__device__ inline unsigned short f2bf(float f) {
  union { float f; unsigned u; } x; x.f = f;
  unsigned r = x.u + 0x7FFFu + ((x.u >> 16) & 1u);   // round-to-nearest-even
  return (unsigned short)(r >> 16);
}

__device__ inline float sigm(float x) { return 1.0f / (1.0f + __expf(-x)); }

// Per-step sync across the 16 workgroups of a (dir, batch-tile) group.
// s_barrier_signal/-wait -3 is the CDNA5 cluster barrier (NOP when the
// dispatch is not clustered); the monotonic global counter is the portable
// fallback that is correct under a plain dispatch with all 256 blocks
// co-resident (256 blocks x 4 waves on >=128 WGPs).
__device__ inline void group_barrier(unsigned* ctr, unsigned target) {
  __builtin_amdgcn_s_cluster_barrier();
  __syncthreads();
  if (threadIdx.x == 0) {
    __builtin_amdgcn_fence(__ATOMIC_RELEASE, "agent");
    __hip_atomic_fetch_add(ctr, 1u, __ATOMIC_RELAXED, __HIP_MEMORY_SCOPE_AGENT);
    while (__hip_atomic_load(ctr, __ATOMIC_RELAXED, __HIP_MEMORY_SCOPE_AGENT) < target)
      __builtin_amdgcn_s_sleep(8);
    __builtin_amdgcn_fence(__ATOMIC_ACQUIRE, "agent");
  }
  __syncthreads();
}

__global__ void __launch_bounds__(THREADS, 2)
bilstm_kernel(const float* __restrict__ seq,            // [B,T,3] f32
              const unsigned char* __restrict__ maskp,  // [B,T] bool (1B)
              const float* __restrict__ Wx_f, const float* __restrict__ Wh_f,
              const float* __restrict__ b_f,
              const float* __restrict__ Wx_b, const float* __restrict__ Wh_b,
              const float* __restrict__ b_b,
              float* __restrict__ out,
              unsigned short* __restrict__ hs,          // [2][2][128][512] bf16
              unsigned* __restrict__ ctrs)
{
  const int bx     = blockIdx.x;
  const int dir    = bx >> 7;
  const int btile  = (bx >> 4) & 7;
  const int ublock = bx & 15;
  const int group  = bx >> 4;
  const int tid    = threadIdx.x;
  const int wave   = tid >> 5;
  const int lane   = tid & 31;
  const int b0     = btile * BTILE;
  const int u0     = ublock * UBLK;

  const float* Wx = dir ? Wx_b : Wx_f;
  const float* Wh = dir ? Wh_b : Wh_f;
  const float* bv = dir ? b_b  : b_f;
  unsigned* ctr = (unsigned*)((char*)ctrs + group * WS_CTR_STRIDE);

  extern __shared__ char smem[];
  unsigned short* whfrag = (unsigned short*)smem;          // [4][2][16][32][16] bf16
  unsigned short* htile  = whfrag + 4*2*16*32*16;          // [16][512] bf16
  float* zbuf   = (float*)(htile + BTILE*UNITS);           // [4][16][32] f32 gates
  float* ctile  = zbuf + 4*BTILE*UBLK;                     // [16][32] cell state
  float* hftile = ctile + BTILE*UBLK;                      // [16][32] hidden f32
  float* wxl    = hftile + BTILE*UBLK;                     // [3][128] Wx cols + [128] bias
  float* xt     = wxl + 4*128;                             // [16][3] x_t

  // ---- One-time fills: Wh slice pre-swizzled into WMMA B-fragment layout.
  // B (32x16, contiguous k order, same as A): lane n<16 -> col n, K=0..15; lane 16+n -> col n, K=16..31.
  for (int c = tid; c < 4096; c += THREADS) {
    const int w    = c >> 10;
    const int tile = (c >> 9) & 1;
    const int kk   = (c >> 5) & 15;
    const int ln   = c & 31;
    const int coll  = ln & 15;
    const int kbase = kk * 32 + (ln >> 4) * 16;
    const int gcol  = w * UNITS + u0 + tile * 16 + coll;
    unsigned short* dst = whfrag + c * 16;
    #pragma unroll 4
    for (int h = 0; h < 16; ++h)
      dst[h] = f2bf(Wh[(size_t)(kbase + h) * GATES + gcol]);
  }
  for (int i = tid; i < 3*128; i += THREADS) {
    const int k = i >> 7, cl = i & 127;
    wxl[i] = Wx[(size_t)k * GATES + (cl >> 5) * UNITS + u0 + (cl & 31)];
  }
  for (int i = tid; i < 128; i += THREADS)
    wxl[384 + i] = bv[(i >> 5) * UNITS + u0 + (i & 31)];
  for (int e = tid; e < BTILE*UBLK; e += THREADS) { ctile[e] = 0.f; hftile[e] = 0.f; }

  // Zero our slice of the h exchange buffer (h0 = 0), buffer 0.
  for (int e = tid; e < BTILE*UBLK; e += THREADS) {
    const int r = e >> 5, u = e & 31;
    hs[((size_t)(0*2 + dir)*BATCH + b0 + r) * UNITS + u0 + u] = 0;
  }

  unsigned epoch = 1;
  group_barrier(ctr, NBLK_GROUP * epoch); ++epoch;

  for (int s = 0; s < TSTEPS; ++s) {
    const int t   = dir ? (TSTEPS - 1 - s) : s;
    const int cur = s & 1, nxt = cur ^ 1;

    // ---- Stage h_{s} tile [16][512] bf16 into LDS (from L2-resident buffer).
    const unsigned short* hsrc = hs + ((size_t)(cur*2 + dir)*BATCH + b0) * UNITS;
    for (int idx = tid; idx < (BTILE*UNITS)/8; idx += THREADS) {
      const int r = idx >> 6, c8 = (idx & 63) * 8;
      *(uint4*)(htile + r*UNITS + c8) = *(const uint4*)(hsrc + (size_t)r*UNITS + c8);
    }
    if (tid < BTILE*3) {
      const int r = tid / 3, k = tid % 3;
      xt[tid] = seq[((size_t)(b0 + r) * TSTEPS + t) * 3 + k];
    }
    __syncthreads();

    // ---- Recurrent GEMM: 16x16 tiles, K=512 in 16 chunks of 32 (bf16 WMMA).
    // A (16x32 bf16): lane l -> row l&15; halfwords K = off..off+15.
    v8f acc0 = {}; v8f acc1 = {};
    const int arow = lane & 15;
    const int aoff = (lane < 16) ? 0 : 16;
    #pragma unroll 4
    for (int kk = 0; kk < 16; ++kk) {
      Frag A, B0, B1;
      const unsigned short* ap = htile + arow*UNITS + kk*32 + aoff;
      A.p.lo = *(const ushort8*)ap;
      A.p.hi = *(const ushort8*)(ap + 8);
      const unsigned short* bp0 = whfrag + (((wave*2 + 0)*16 + kk)*32 + lane)*16;
      const unsigned short* bp1 = whfrag + (((wave*2 + 1)*16 + kk)*32 + lane)*16;
      B0.p.lo = *(const ushort8*)bp0; B0.p.hi = *(const ushort8*)(bp0 + 8);
      B1.p.lo = *(const ushort8*)bp1; B1.p.hi = *(const ushort8*)(bp1 + 8);
      acc0 = __builtin_amdgcn_wmma_f32_16x16x32_bf16(false, A.v, false, B0.v,
                                                     (short)0, acc0, false, false);
      acc1 = __builtin_amdgcn_wmma_f32_16x16x32_bf16(false, A.v, false, B1.v,
                                                     (short)0, acc1, false, false);
    }

    // ---- Spill gate pre-activations: C layout lane l, vgpr v -> row v + 8*(l>=16), col l&15.
    {
      const int zr = (lane < 16) ? 0 : 8;
      const int zc = lane & 15;
      float* zb = zbuf + wave * (BTILE*UBLK);
      #pragma unroll
      for (int v = 0; v < 8; ++v) {
        zb[(zr + v)*UBLK + zc]       = acc0[v];
        zb[(zr + v)*UBLK + 16 + zc]  = acc1[v];
      }
    }
    __syncthreads();

    // ---- Pointwise: add x@Wx + b, gates, cell update, mask carry, outputs.
    for (int e = tid; e < BTILE*UBLK; e += THREADS) {
      const int r = e >> 5, u = e & 31;
      const float x0 = xt[r*3+0], x1 = xt[r*3+1], x2 = xt[r*3+2];
      auto zg = [&](int gt) -> float {
        const int cl = gt*32 + u;
        return zbuf[gt*(BTILE*UBLK) + e] + wxl[384 + cl]
             + x0*wxl[cl] + x1*wxl[128 + cl] + x2*wxl[256 + cl];
      };
      const float ig = sigm(zg(0));
      const float fg = sigm(zg(1));
      const float gg = tanhf(zg(2));
      const float og = sigm(zg(3));
      const float cn = fg * ctile[e] + ig * gg;
      const float hn = og * tanhf(cn);
      const bool  m  = maskp[(size_t)(b0 + r) * TSTEPS + t] != 0;
      const float c2 = m ? cn : ctile[e];
      const float h2 = m ? hn : hftile[e];
      ctile[e] = c2; hftile[e] = h2;
      const int gb = b0 + r, gu = u0 + u;
      out[((size_t)gb * TSTEPS + t) * 1024 + dir * UNITS + gu] = h2;       // ys
      hs[((size_t)(nxt*2 + dir)*BATCH + gb) * UNITS + gu] = f2bf(h2);      // h_{s+1}
    }

    group_barrier(ctr, NBLK_GROUP * epoch); ++epoch;
  }

  // ---- Final states: state_h / state_c [B,2,U].
  for (int e = tid; e < BTILE*UBLK; e += THREADS) {
    const int r = e >> 5, u = e & 31;
    const size_t o = ((size_t)(b0 + r)*2 + dir)*UNITS + u0 + u;
    out[SH_OFF + o] = hftile[e];
    out[SC_OFF + o] = ctile[e];
  }
}

// pos_agg = (state_h @ W1 + b1).reshape(B,1,8) — 1024 length-512 dots.
__global__ void posagg_kernel(const float* __restrict__ W1,
                              const float* __restrict__ b1,
                              float* __restrict__ out)
{
  const int g = blockIdx.x * blockDim.x + threadIdx.x;
  if (g >= 1024) return;
  const int b = g >> 3, d = (g >> 2) & 1, j = g & 3;
  const float* h = out + SH_OFF + ((size_t)b*2 + d) * UNITS;
  float s = b1[j];
  for (int u = 0; u < UNITS; ++u) s += h[u] * W1[u*4 + j];
  out[PA_OFF + (size_t)b*8 + d*4 + j] = s;
}

extern "C" void kernel_launch(void* const* d_in, const int* in_sizes, int n_in,
                              void* d_out, int out_size, void* d_ws, size_t ws_size,
                              hipStream_t stream) {
  const float*         seq   = (const float*)d_in[0];
  const unsigned char* maskp = (const unsigned char*)d_in[1];   // jnp bool = 1 byte
  const float* Wx_f = (const float*)d_in[2];
  const float* Wh_f = (const float*)d_in[3];
  const float* b_f  = (const float*)d_in[4];
  const float* Wx_b = (const float*)d_in[5];
  const float* Wh_b = (const float*)d_in[6];
  const float* b_b  = (const float*)d_in[7];
  const float* W1   = (const float*)d_in[8];
  const float* b1   = (const float*)d_in[9];
  float* out = (float*)d_out;

  unsigned*       ctrs = (unsigned*)d_ws;
  unsigned short* hs   = (unsigned short*)((char*)d_ws + WS_HS_OFF);

  // Barrier counters must start at 0 every call (deterministic, capture-safe).
  hipMemsetAsync(d_ws, 0, WS_HS_OFF, stream);

  // ~158 KB dynamic LDS per block (320 KB/WGP on CDNA5) — opt in every call.
  hipFuncSetAttribute(reinterpret_cast<const void*>(bilstm_kernel),
                      hipFuncAttributeMaxDynamicSharedMemorySize, SMEM_BYTES);

  hipLaunchKernelGGL(bilstm_kernel, dim3(256), dim3(THREADS), SMEM_BYTES, stream,
                     seq, maskp, Wx_f, Wh_f, b_f, Wx_b, Wh_b, b_b, out, hs, ctrs);
  hipLaunchKernelGGL(posagg_kernel, dim3(4), dim3(256), 0, stream, W1, b1, out);
}
